// LSTMLayerNorm_62766652063947
// MI455X (gfx1250) — matrix operand order repaired and checked by reading, not compiled, not run on hardware
//
#include <hip/hip_runtime.h>
#include <hip/hip_bf16.h>
#include <stdint.h>

#define NL 2
#define DD 512
#define BB 32
#define UU 2048
#define FD 2048  /* 4*D */
#define EPS 1e-3f

typedef __attribute__((ext_vector_type(16))) _Float16 v16h;
typedef __attribute__((ext_vector_type(8)))  _Float16 v8h;
typedef __attribute__((ext_vector_type(8)))  float    v8f;

union AB { v16h v; v8h h[2]; };

__device__ inline v8f wmma_f16(v16h a, v16h b, v8f c) {
  // D = A(16x32 f16) * B(32x16 f16) + C(16x16 f32)
  return __builtin_amdgcn_wmma_f32_16x16x32_f16(false, a, false, b, (short)0, c,
                                                false, false);
}

__device__ inline float sigm(float x) { return 1.0f / (1.0f + expf(-x)); }

__device__ inline float2 blockReduce2(float a, float b, float* r1, float* r2) {
  const int tid = threadIdx.x;
  r1[tid] = a; r2[tid] = b;
  __syncthreads();
  for (int s = 128; s > 0; s >>= 1) {
    if (tid < s) { r1[tid] += r1[tid + s]; r2[tid] += r2[tid + s]; }
    __syncthreads();
  }
  float2 res = make_float2(r1[0], r2[0]);
  __syncthreads();
  return res;
}

// Grid-wide barrier (all WGs co-resident: grid = 32 WGs).
__device__ inline void gridbar(int* cnt, int* flag, int tag) {
  __syncthreads();
  if (threadIdx.x == 0) {
    __threadfence();                       // release
    int prev = atomicAdd(cnt, 1);
    if (prev == (int)gridDim.x - 1) {
      atomicExch(cnt, 0);
      __threadfence();
      atomicExch(flag, tag);
    } else {
      while (atomicAdd(flag, 0) < tag) { __builtin_amdgcn_s_sleep(2); }
    }
  }
  __syncthreads();
  __threadfence();                         // acquire for every thread
}

// ---------------- Weight transpose+convert: W[l][k][n] f32 -> WT[l][n][k] f16
__global__ void k_cvt_w(const float* __restrict__ Wk, const float* __restrict__ Wr,
                        _Float16* __restrict__ wkT, _Float16* __restrict__ wrT) {
  long idx = (long)blockIdx.x * blockDim.x + threadIdx.x;
  if (idx >= (long)2 * NL * FD * DD) return;
  int k = (int)(idx % DD); long t = idx / DD;
  int n = (int)(t % FD);   t /= FD;
  int layer = (int)(t % NL);
  int mat   = (int)(t / NL);
  const float* src = mat ? Wr : Wk;
  _Float16*    dst = mat ? wrT : wkT;
  dst[((long)layer * FD + n) * DD + k] =
      (_Float16)src[((long)layer * DD + k) * FD + n];
}

// ---------------- zraw = x @ Wk   (16 rows x 2048 cols per block, WMMA f16)
__global__ __launch_bounds__(256) void k_gemm_zx(const float* __restrict__ x,
                                                 const _Float16* __restrict__ wkT,
                                                 _Float16* __restrict__ zraw) {
  __shared__ _Float16 As[16][520];     // 16 rows x 512 K, padded
  const int tid = threadIdx.x;
  const long r0 = (long)blockIdx.x * 16;
  for (int i = 0; i < 32; ++i) {       // 16*512 f32 -> f16 LDS
    int e = i * 256 + tid;
    int row = e >> 9, k = e & 511;
    As[row][k] = (_Float16)x[(r0 + row) * DD + k];
  }
  __syncthreads();
  const int lane = tid & 31, wv = tid >> 5;
  const int lr = lane & 15, hi = lane >> 4;
  for (int nt = 0; nt < 16; ++nt) {
    const int n0 = (nt * 8 + wv) * 16;
    v8f acc = {};
    for (int kc = 0; kc < 16; ++kc) {
      AB a, b;
      const int kbA = kc * 32 + hi * 8;          // A: K {0..7,16..23}/{8..15,24..31}
      a.h[0] = *(const v8h*)&As[lr][kbA];
      a.h[1] = *(const v8h*)&As[lr][kbA + 16];
      const int kbB = kc * 32 + hi * 8;          // B: K {0..7,16..23}/{8..15,24..31}
      const _Float16* bp = wkT + (long)(n0 + lr) * DD + kbB;
      b.h[0] = *(const v8h*)bp;
      b.h[1] = *(const v8h*)(bp + 16);
      acc = wmma_f16(a.v, b.v, acc);
    }
    for (int r = 0; r < 8; ++r)                  // D: M = r + 8*hi, N = lr
      zraw[(r0 + r + hi * 8) * (long)FD + n0 + lr] = (_Float16)acc[r];
  }
}

// ---------------- in-place row layernorm over 2048 (one block per row)
__global__ __launch_bounds__(256) void k_rowln(_Float16* __restrict__ z,
                                               const float* __restrict__ g,
                                               const float* __restrict__ bb) {
  __shared__ float r1[256], r2[256];
  const int tid = threadIdx.x;
  const long base = (long)blockIdx.x * FD;
  float v[8], s = 0.f, sq = 0.f;
  for (int i = 0; i < 8; ++i) {
    v[i] = (float)z[base + i * 256 + tid];
    s += v[i]; sq += v[i] * v[i];
  }
  float2 red = blockReduce2(s, sq, r1, r2);
  float mean = red.x * (1.0f / FD);
  float rs = rsqrtf(red.y * (1.0f / FD) - mean * mean + EPS);
  for (int i = 0; i < 8; ++i) {
    int col = i * 256 + tid;
    z[base + col] = (_Float16)(g[col] * (v[i] - mean) * rs + bb[col]);
  }
}

// ---------------- persistent cooperative LSTM scan (32 WGs x 256 thr)
__global__ __launch_bounds__(256) void k_scan(
    const _Float16* __restrict__ wrT, const _Float16* __restrict__ zx,
    const float* __restrict__ h0, const float* __restrict__ c0,
    const float* __restrict__ bias, const float* __restrict__ rg,
    const float* __restrict__ rb, const float* __restrict__ sg,
    const float* __restrict__ sb, const int* __restrict__ lens,
    float* __restrict__ outp, float* __restrict__ hT, float* __restrict__ cT,
    _Float16* __restrict__ hbuf, float* __restrict__ zbuf,
    int* __restrict__ barCnt, int* __restrict__ barFlag) {
  __shared__ _Float16 Bsl[64][520];              // persistent Wr^T slice (64 cols)
  __shared__ float c_row[512], h_row[512], cn[512], ow[512];
  __shared__ float r1[256], r2[256];
  const int tid = threadIdx.x, wg = blockIdx.x;
  const int lane = tid & 31, wv = tid >> 5;
  const int lr = lane & 15, hi = lane >> 4;

  for (int it = 0; it < 16; ++it) {              // load Wr slice once
    int chunk = it * 256 + tid;                  // 64 rows * 64 v8h chunks
    int n = chunk >> 6, k8 = chunk & 63;
    *(v8h*)&Bsl[n][k8 * 8] = *(const v8h*)(wrT + ((long)(wg * 64 + n)) * DD + k8 * 8);
  }
  for (int j = 0; j < 2; ++j) {                  // init state row `wg`
    int d = tid + j * 256;
    c_row[d] = c0[wg * DD + d];
    float h = h0[wg * DD + d];
    h_row[d] = h;
    hbuf[wg * DD + d] = (_Float16)h;
  }
  const int len = lens[wg];
  __syncthreads();
  int tag = 0;
  gridbar(barCnt, barFlag, ++tag);               // h f16 visible everywhere

  const int m0 = (wv >> 2) * 16;                 // 2 M-tiles x 4 N-tiles over 8 waves
  const int nl0 = (wv & 3) * 16;
  const int ncol = wg * 64 + nl0 + lr;

  for (int t = 0; t < UU; ++t) {
    // ---- stage 1: y[:, wg*64 : wg*64+64] = h @ Wr_slice ----
    v8f acc = {};
    for (int kc = 0; kc < 16; ++kc) {
      AB a, b;
      const int kbA = kc * 32 + hi * 8;
      const _Float16* ap = hbuf + (m0 + lr) * DD + kbA;
      a.h[0] = *(const v8h*)ap;
      a.h[1] = *(const v8h*)(ap + 16);
      const int kbB = kc * 32 + hi * 8;
      const _Float16* bp = &Bsl[nl0 + lr][kbB];
      b.h[0] = *(const v8h*)bp;
      b.h[1] = *(const v8h*)(bp + 16);
      acc = wmma_f16(a.v, b.v, acc);
    }
    for (int r = 0; r < 8; ++r)
      zbuf[(m0 + r + hi * 8) * FD + ncol] = acc[r];
    gridbar(barCnt, barFlag, ++tag);

    // ---- stage 2: WG `wg` owns batch row `wg` entirely ----
    float s = 0.f, sq = 0.f;
    for (int i = 0; i < 8; ++i) {
      float v = zbuf[wg * FD + i * 256 + tid];
      s += v; sq += v * v;
    }
    float2 red = blockReduce2(s, sq, r1, r2);
    float mean = red.x * (1.0f / FD);
    float rs = rsqrtf(red.y * (1.0f / FD) - mean * mean + EPS);
    const _Float16* zxrow = zx + ((long)wg * UU + t) * FD;
    for (int j = 0; j < 2; ++j) {
      int d = tid + j * 256;
      float zv[4];
      for (int q = 0; q < 4; ++q) {
        int col = q * 512 + d;
        float y = zbuf[wg * FD + col];
        float yn = rg[col] * (y - mean) * rs + rb[col];
        zv[q] = (float)zxrow[col] + yn + bias[col];
      }
      float iv = sigm(zv[0]), fv = sigm(zv[1]);
      float gv = tanhf(zv[2]), ov = sigm(zv[3]);
      cn[d] = fv * c_row[d] + iv * gv;
      ow[d] = ov;
    }
    __syncthreads();
    float s2 = 0.f, q2 = 0.f;
    for (int j = 0; j < 2; ++j) { float v = cn[tid + j * 256]; s2 += v; q2 += v * v; }
    float2 red2 = blockReduce2(s2, q2, r1, r2);
    float m2 = red2.x * (1.0f / DD);
    float rs2 = rsqrtf(red2.y * (1.0f / DD) - m2 * m2 + EPS);
    const bool msk = (t < len);
    for (int j = 0; j < 2; ++j) {
      int d = tid + j * 256;
      float cl = sg[d] * (cn[d] - m2) * rs2 + sb[d];
      float hn = ow[d] * tanhf(cl);
      if (msk) { c_row[d] = cl; h_row[d] = hn; }
      float h = h_row[d];
      outp[((long)wg * UU + t) * DD + d] = h;
      hbuf[wg * DD + d] = (_Float16)h;
    }
    gridbar(barCnt, barFlag, ++tag);             // h visible before next step
  }
  for (int j = 0; j < 2; ++j) {
    int d = tid + j * 256;
    hT[wg * DD + d] = h_row[d];
    cT[wg * DD + d] = c_row[d];
  }
}

extern "C" void kernel_launch(void* const* d_in, const int* in_sizes, int n_in,
                              void* d_out, int out_size, void* d_ws, size_t ws_size,
                              hipStream_t stream) {
  const float* inputs = (const float*)d_in[0];
  const int*   lens   = (const int*)d_in[1];
  const float* states = (const float*)d_in[2];
  const float* Wk     = (const float*)d_in[3];
  const float* Wr     = (const float*)d_in[4];
  const float* biases = (const float*)d_in[5];
  const float* kng    = (const float*)d_in[6];
  const float* knb    = (const float*)d_in[7];
  const float* rng    = (const float*)d_in[8];
  const float* rnb    = (const float*)d_in[9];
  const float* sng    = (const float*)d_in[10];
  const float* snb    = (const float*)d_in[11];
  float* out = (float*)d_out;
  float* statesOut = out + (long)BB * UU * DD;

  char* ws = (char*)d_ws;
  _Float16* wkT  = (_Float16*)ws;                       // NL*FD*DD f16 (4 MB)
  _Float16* wrT  = wkT + (long)NL * FD * DD;            // NL*FD*DD f16 (4 MB)
  _Float16* zraw = wrT + (long)NL * FD * DD;            // BB*UU*FD f16 (256 MB)
  float*    zbuf = (float*)(zraw + (long)BB * UU * FD); // BB*FD f32 (256 KB)
  _Float16* hbuf = (_Float16*)(zbuf + BB * FD);         // BB*DD f16 (32 KB)
  int* bar = (int*)(((uintptr_t)(hbuf + BB * DD) + 255) & ~(uintptr_t)255);

  k_cvt_w<<<(2 * NL * FD * DD + 255) / 256, 256, 0, stream>>>(Wk, Wr, wkT, wrT);

  for (int layer = 0; layer < NL; ++layer) {
    const float* x = (layer == 0) ? inputs : out;  // layer-0 outputs live in d_out
    k_gemm_zx<<<(BB * UU) / 16, 256, 0, stream>>>(x, wkT + (long)layer * FD * DD, zraw);
    k_rowln<<<BB * UU, 256, 0, stream>>>(zraw, kng + layer * FD, knb + layer * FD);
    hipMemsetAsync(bar, 0, 256, stream);
    k_scan<<<BB, 256, 0, stream>>>(
        wrT + (long)layer * FD * DD, zraw,
        states + (long)(layer * 2 + 0) * BB * DD,
        states + (long)(layer * 2 + 1) * BB * DD,
        biases + layer * FD, rng + layer * FD, rnb + layer * FD,
        sng + layer * DD, snb + layer * DD, lens,
        out,
        statesOut + (long)(layer * 2 + 0) * BB * DD,
        statesOut + (long)(layer * 2 + 1) * BB * DD,
        hbuf, zbuf, bar, bar + 32);
  }
}
